// SpanIndexEncoder_31129922962204
// MI455X (gfx1250) — matrix operand order repaired and checked by reading, not compiled, not run on hardware
//
#include <hip/hip_runtime.h>

// SpanIndexEncoder as an on-the-fly-mask GEMM on CDNA5 WMMA.
//   out[t,f] = sum_n (start[n] <= t <= end[n]) * embed[n,f]
//   T = 2048 tokens, N = 1024 nodes, F = 128 features.
//
// Phase 1 (embed_split_kernel): embed f32 -> hi/lo f16 planes in d_ws, once.
//   hi = f16(x), lo = f16(x - hi); mask entries are exact {0,1} in f16 and
//   1.0*hi / 1.0*lo accumulate exactly in f32, so the result matches the f32
//   reference to ~2^-22 relative error at 2x the (negligible) WMMA count.
// Phase 2 (span_wmma_kernel): one wave32 computes a 16(T) x 32(F) f32 tile.
//   Per 32-wide K step: one register-built mask A feeds 4x
//   v_wmma_f32_16x16x32_f16 (2 F-subtiles x hi/lo). B operands are direct
//   32-byte loads from the pre-split planes (no conversion VALU in the loop).

typedef __attribute__((ext_vector_type(16))) _Float16     v16h;
typedef __attribute__((ext_vector_type(8)))  float        v8f;
typedef __attribute__((ext_vector_type(8)))  unsigned int v8u;

#define TOK 2048
#define NND 1024
#define FEA 128

__global__ __launch_bounds__(256) void embed_split_kernel(
    const float* __restrict__ embed,
    _Float16*    __restrict__ wsH,
    _Float16*    __restrict__ wsL)
{
    const int idx = blockIdx.x * 256 + threadIdx.x;   // 0 .. 131071
    const float    x = embed[idx];
    const _Float16 h = (_Float16)x;
    wsH[idx] = h;
    wsL[idx] = (_Float16)(x - (float)h);
}

template <bool USE_WS>
__global__ __launch_bounds__(128) void span_wmma_kernel(
    const int*      __restrict__ starts,
    const int*      __restrict__ ends,
    const float*    __restrict__ embed,
    const _Float16* __restrict__ wsH,
    const _Float16* __restrict__ wsL,
    float*          __restrict__ out)
{
    // Stage span bounds once per block: 1024 x (start,end) = 8 KB of LDS.
    __shared__ int2 se[NND];
    const int tid = threadIdx.x;
    for (int i = tid; i < NND; i += 128) {
        int2 v; v.x = starts[i]; v.y = ends[i];
        se[i] = v;
    }
    __syncthreads();

    const int wave  = tid >> 5;                 // 4 waves / block
    const int lane  = tid & 31;
    const int g     = blockIdx.x * 4 + wave;    // 0..511 macro-tiles (16T x 32F)
    const int tt    = g >> 2;                   // T tile: 0..127
    const int fbase = (g & 3) * 32;             // F base: 0,32,64,96

    // A-matrix row this lane owns (ISA 16-bit A 16x32 layout): M = lane&15.
    const int t   = tt * 16 + (lane & 15);
    const int hiK = (lane & 16) ? 8 : 0;        // lanes 16..31 hold the +8 K half

    v8f acc0 = {};                              // columns fbase .. fbase+15
    v8f acc1 = {};                              // columns fbase+16 .. fbase+31

    for (int k0 = 0; k0 < NND; k0 += 32) {
        // ---- Build A: 16x32 f16 mask tile, two packed halves per u32 ----
        // element j (0..15) of this lane maps to K = hiK + (j<8 ? j : j+8)
        v8u a_pk;
        #pragma unroll
        for (int v = 0; v < 8; ++v) {
            const int j0 = 2 * v, j1 = 2 * v + 1;
            const int K0 = hiK + (j0 < 8 ? j0 : j0 + 8);
            const int K1 = hiK + (j1 < 8 ? j1 : j1 + 8);
            const int2 s0 = se[k0 + K0];        // ds_load_b64, lane-broadcast
            const int2 s1 = se[k0 + K1];
            const unsigned int b0 = (s0.x <= t && t <= s0.y) ? 0x3C00u : 0u; // f16 1.0
            const unsigned int b1 = (s1.x <= t && t <= s1.y) ? 0x3C00u : 0u;
            a_pk[v] = b0 | (b1 << 16);
        }
        const v16h A = __builtin_bit_cast(v16h, a_pk);

        // ---- B operands: lane&15 = F column, elements = K (same order as A) ----
        v16h Bh0, Bl0, Bh1, Bl1;
        if constexpr (USE_WS) {
            const size_t ro = (size_t)(k0 + lane) * FEA + fbase;
            if (k0 + 32 < NND) {  // speculative prefetch of next K rows
                __builtin_prefetch(wsH + ro + 32 * FEA, 0, 0);
                __builtin_prefetch(wsL + ro + 32 * FEA, 0, 0);
            }
            #pragma unroll
            for (int i = 0; i < 16; ++i) {
                const size_t o = (size_t)(k0 + hiK + (i < 8 ? i : i + 8)) * FEA + fbase + (lane & 15);
                Bh0[i] = wsH[o];  Bh1[i] = wsH[o + 16];
                Bl0[i] = wsL[o];  Bl1[i] = wsL[o + 16];
            }
        } else {
            // Fallback (ws too small): split f32 -> hi/lo f16 inline.
            const float* rp = embed + fbase + (lane & 15);
            #pragma unroll
            for (int i = 0; i < 16; ++i) {
                const float*   q  = rp + (size_t)(k0 + hiK + (i < 8 ? i : i + 8)) * FEA;
                const float    x0 = q[0],           x1 = q[16];
                const _Float16 h0 = (_Float16)x0,   h1 = (_Float16)x1;
                Bh0[i] = h0;  Bl0[i] = (_Float16)(x0 - (float)h0);
                Bh1[i] = h1;  Bl1[i] = (_Float16)(x1 - (float)h1);
            }
        }

        // ---- 4 WMMAs per K step: one A, two F-subtiles x (lo, hi) ----
        acc0 = __builtin_amdgcn_wmma_f32_16x16x32_f16(
                   false, A, false, Bl0, (short)0, acc0, false, false);
        acc0 = __builtin_amdgcn_wmma_f32_16x16x32_f16(
                   false, A, false, Bh0, (short)0, acc0, false, false);
        acc1 = __builtin_amdgcn_wmma_f32_16x16x32_f16(
                   false, A, false, Bl1, (short)0, acc1, false, false);
        acc1 = __builtin_amdgcn_wmma_f32_16x16x32_f16(
                   false, A, false, Bh1, (short)0, acc1, false, false);
    }

    // ---- Store C/D: VGPR r -> row r + (lane>=16 ? 8 : 0), col lane&15 ----
    const int mHi = (lane & 16) ? 8 : 0;
    const int n   = lane & 15;
    #pragma unroll
    for (int r = 0; r < 8; ++r) {
        const int row = tt * 16 + r + mHi;
        float* op = out + (size_t)row * FEA + fbase + n;
        op[0]  = acc0[r];
        op[16] = acc1[r];
    }
}

extern "C" void kernel_launch(void* const* d_in, const int* in_sizes, int n_in,
                              void* d_out, int out_size, void* d_ws, size_t ws_size,
                              hipStream_t stream) {
    const int*   starts = (const int*)d_in[0];   // node_span_starts [1024]
    const int*   ends   = (const int*)d_in[1];   // node_span_ends   [1024]
    const float* embed  = (const float*)d_in[2]; // embed [1024,128] f32
    float*       out    = (float*)d_out;         // out [2048,128] f32

    const size_t WS_NEED = (size_t)NND * FEA * sizeof(_Float16) * 2;  // 512 KB
    if (ws_size >= WS_NEED) {
        _Float16* wsH = (_Float16*)d_ws;
        _Float16* wsL = wsH + (size_t)NND * FEA;
        embed_split_kernel<<<dim3((NND * FEA) / 256), dim3(256), 0, stream>>>(embed, wsH, wsL);
        span_wmma_kernel<true><<<dim3(128), dim3(128), 0, stream>>>(
            starts, ends, embed, wsH, wsL, out);
    } else {
        span_wmma_kernel<false><<<dim3(128), dim3(128), 0, stream>>>(
            starts, ends, embed, nullptr, nullptr, out);
    }
}
